// FormulaNet_30133490549458
// MI455X (gfx1250) — compile-verified, matrix operand order checked by reading, not run on hardware
//
#include <hip/hip_runtime.h>
#include <hip/hip_bf16.h>
#include <cstdint>
#include <cstddef>

// ---------------------------------------------------------------------------
// FormulaNet (GAT-like GNN) for gfx1250.
//  - GEMMs: bf16 x bf16 -> f32 via v_wmma_f32_16x16x32_bf16 (wave32 WMMA).
//  - Edge softmax/scatter: L2-resident atomics (xs/agg ~25.6MB each << 192MB L2).
//  - Training-mode BN: bias-before-BN cancels exactly -> folded out.
// ---------------------------------------------------------------------------

typedef __attribute__((ext_vector_type(16))) __bf16 v16bf;
typedef __attribute__((ext_vector_type(8)))  float  v8f;

#define DEVINL __device__ __forceinline__

constexpr int NN   = 50000;
constexpr int EE   = 800000;
constexpr int FIN  = 92;
constexpr int FPAD = 96;      // 92 padded to multiple of 32? -> 96 (3 K-chunks of 32)
constexpr int HID  = 64;
constexpr int HH   = 2;
constexpr int LL   = 4;
constexpr int GG   = 256;
constexpr int C2   = HH * HID;   // 128
constexpr int ET   = EE + NN;    // edges incl. self loops
constexpr float BN_EPS = 1e-5f;

DEVINL __bf16 f2bf(float f) {
  unsigned u = __builtin_bit_cast(unsigned, f);
  unsigned r = (u + 0x7FFFu + ((u >> 16) & 1u)) >> 16;   // round-to-nearest-even
  unsigned short s = (unsigned short)r;
  return __builtin_bit_cast(__bf16, s);
}
// monotonic float<->uint encoding for atomicMax on floats (handles negatives)
DEVINL unsigned encf(float f) {
  unsigned u = __builtin_bit_cast(unsigned, f);
  return (u & 0x80000000u) ? ~u : (u | 0x80000000u);
}
DEVINL float decf(unsigned e) {
  unsigned u = (e & 0x80000000u) ? (e & 0x7FFFFFFFu) : ~e;
  return __builtin_bit_cast(float, u);
}

// ---------------------------------------------------------------------------
// fill / convert helpers
// ---------------------------------------------------------------------------
__global__ void fill_u32_kernel(unsigned* __restrict__ p, size_t n, unsigned v) {
  size_t i = (size_t)blockIdx.x * blockDim.x + threadIdx.x;
  if (i < n) p[i] = v;
}

// convert f32 [rows, scols] -> bf16 [rows, dcols], zero-padding cols >= scols
__global__ void cvt_pad_bf16_kernel(const float* __restrict__ src, __bf16* __restrict__ dst,
                                    int rows, int scols, int dcols) {
  size_t i = (size_t)blockIdx.x * blockDim.x + threadIdx.x;
  size_t total = (size_t)rows * dcols;
  if (i >= total) return;
  int r = (int)(i / dcols), c = (int)(i % dcols);
  float v = (c < scols) ? src[(size_t)r * scols + c] : 0.0f;
  dst[i] = f2bf(v);
}

// ---------------------------------------------------------------------------
// WMMA GEMM: out[N, MOUT] = A_bf16[N, lda] * W_bf16[MOUT, ldw]^T
// One wave -> 16 rows x MOUT cols.  KCH chunks of K=32.
// MODE 0: out = acc.  MODE 1: out = acc + res (residual), also writes bf16 copy.
// ---------------------------------------------------------------------------
template <int MOUT, int KCH, int MODE>
__global__ void wmma_gemm_kernel(const __bf16* __restrict__ A, int lda,
                                 const __bf16* __restrict__ W, int ldw,
                                 float* __restrict__ out,
                                 const float* __restrict__ res,
                                 __bf16* __restrict__ outbf,
                                 int nrows) {
  const int lane = threadIdx.x & 31;
  const int wave = threadIdx.x >> 5;
  const int tile = blockIdx.x * (blockDim.x >> 5) + wave;
  const int row0 = tile * 16;
  if (row0 >= nrows) return;
  const int half = lane >> 4;   // K-group select (ISA 16-bit A/B layout)
  const int mn   = lane & 15;   // row index for A, col index for B

  constexpr int NT = MOUT / 16;
  v8f acc[NT];
#pragma unroll
  for (int t = 0; t < NT; ++t)
#pragma unroll
    for (int r = 0; r < 8; ++r) acc[t][r] = 0.0f;

  for (int kc = 0; kc < KCH; ++kc) {
    const int kk = kc * 32 + half * 8;
    v16bf a;
    const __bf16* arow = A + (size_t)(row0 + mn) * lda + kk;
#pragma unroll
    for (int j = 0; j < 8; ++j) {
      const int kb = (j < 4) ? (j * 2) : (16 + (j - 4) * 2);
      a[2 * j + 0] = arow[kb + 0];
      a[2 * j + 1] = arow[kb + 1];
    }
#pragma unroll
    for (int t = 0; t < NT; ++t) {
      v16bf b;
      const __bf16* wrow = W + (size_t)(t * 16 + mn) * ldw + kk;
#pragma unroll
      for (int j = 0; j < 8; ++j) {
        const int kb = (j < 4) ? (j * 2) : (16 + (j - 4) * 2);
        b[2 * j + 0] = wrow[kb + 0];
        b[2 * j + 1] = wrow[kb + 1];
      }
      acc[t] = __builtin_amdgcn_wmma_f32_16x16x32_bf16(
          false, a, false, b, (short)0, acc[t], false, false);
    }
  }

#pragma unroll
  for (int t = 0; t < NT; ++t) {
#pragma unroll
    for (int r = 0; r < 8; ++r) {
      const int row = row0 + half * 8 + r;   // C/D layout: M = r + 8*half
      const int col = t * 16 + mn;           // N = lane&15
      const size_t idx = (size_t)row * MOUT + col;
      float v = acc[t][r];
      if (MODE == 1) {
        v += res[idx];
        out[idx]   = v;
        outbf[idx] = f2bf(v);
      } else {
        out[idx] = v;
      }
    }
  }
}

// ---------------------------------------------------------------------------
// BatchNorm (training mode, biased variance). Bias-before-BN cancels exactly.
// ---------------------------------------------------------------------------
__global__ void bn_stats_kernel(const float* __restrict__ x, int nrows, int C,
                                float* __restrict__ sum, float* __restrict__ sumsq) {
  __shared__ float s1[256], s2[256];
  const int c = blockIdx.x;
  float a = 0.f, b = 0.f;
  for (int n = blockIdx.y * blockDim.x + threadIdx.x; n < nrows;
       n += gridDim.y * blockDim.x) {
    float v = x[(size_t)n * C + c];
    a += v; b += v * v;
  }
  s1[threadIdx.x] = a; s2[threadIdx.x] = b;
  __syncthreads();
  for (int o = 128; o > 0; o >>= 1) {
    if ((int)threadIdx.x < o) {
      s1[threadIdx.x] += s1[threadIdx.x + o];
      s2[threadIdx.x] += s2[threadIdx.x + o];
    }
    __syncthreads();
  }
  if (threadIdx.x == 0) {
    atomicAdd(&sum[c],   s1[0]);
    atomicAdd(&sumsq[c], s2[0]);
  }
}

__global__ void bn_finalize_kernel(const float* __restrict__ sum, const float* __restrict__ sumsq,
                                   const float* __restrict__ gamma, const float* __restrict__ beta,
                                   float* __restrict__ scale, float* __restrict__ shift,
                                   int C, float invN) {
  int c = blockIdx.x * blockDim.x + threadIdx.x;
  if (c >= C) return;
  float mu  = sum[c] * invN;
  float var = fmaxf(sumsq[c] * invN - mu * mu, 0.0f);
  float s   = gamma[c] * rsqrtf(var + BN_EPS);
  scale[c] = s;
  shift[c] = beta[c] - mu * s;
}

// y = x*scale[c] + shift[c]  (optional relu); writes f32 and/or bf16 outputs
__global__ void bn_apply_kernel(const float* __restrict__ x,
                                const float* __restrict__ scale, const float* __restrict__ shift,
                                int cmask, size_t total, int relu,
                                float* __restrict__ of32, __bf16* __restrict__ obf) {
  size_t i = (size_t)blockIdx.x * blockDim.x + threadIdx.x;
  if (i >= total) return;
  int c = (int)(i & (size_t)cmask);
  float v = x[i] * scale[c] + shift[c];
  if (relu) v = v > 0.f ? v : 0.f;
  if (of32) of32[i] = v;
  if (obf)  obf[i]  = f2bf(v);
}

// ---------------------------------------------------------------------------
// attention dots: a_src/a_dst[n,h] = <xs[n,h,:], att_src/dst[h,:]>
// ---------------------------------------------------------------------------
__global__ void attn_dots_kernel(const float* __restrict__ xs,
                                 const float* __restrict__ attS, const float* __restrict__ attD,
                                 float* __restrict__ as_, float* __restrict__ ad_, int nrows) {
  int i = blockIdx.x * blockDim.x + threadIdx.x;   // i = n*H + h
  if (i >= nrows * HH) return;
  int n = i >> 1, h = i & 1;
  const float* xp = xs + (size_t)n * C2 + h * HID;
  const float* sp = attS + h * HID;
  const float* dp = attD + h * HID;
  float a = 0.f, b = 0.f;
#pragma unroll 8
  for (int c = 0; c < HID; ++c) { float v = xp[c]; a += v * sp[c]; b += v * dp[c]; }
  as_[i] = a; ad_[i] = b;
}

// ---------------------------------------------------------------------------
// edge pass 1: alpha = leaky_relu(a_src[s]+a_dst[d]) * aw[s]; segment max(dst)
// ---------------------------------------------------------------------------
__global__ void edge_alpha_kernel(const int* __restrict__ src, const int* __restrict__ dst,
                                  const float* __restrict__ as_, const float* __restrict__ ad_,
                                  const float* __restrict__ aw,
                                  float* __restrict__ alpha, unsigned* __restrict__ amax) {
  int e = blockIdx.x * blockDim.x + threadIdx.x;
  if (e >= ET) return;
  int s, d;
  if (e < EE) { s = src[e]; d = dst[e]; } else { s = d = e - EE; }
  float w = aw[s];
#pragma unroll
  for (int h = 0; h < HH; ++h) {
    float a = as_[s * HH + h] + ad_[d * HH + h];
    a = a > 0.f ? a : 0.2f * a;        // leaky_relu, slope 0.2
    a *= w;
    alpha[(size_t)e * HH + h] = a;
    atomicMax(&amax[d * HH + h], encf(a));
  }
}

// edge pass 2: ex = exp(alpha - max); segment sum(dst)
__global__ void edge_exp_kernel(const int* __restrict__ dst,
                                const unsigned* __restrict__ amax,
                                float* __restrict__ alpha, float* __restrict__ denom) {
  int e = blockIdx.x * blockDim.x + threadIdx.x;
  if (e >= ET) return;
  int d = (e < EE) ? dst[e] : (e - EE);
#pragma unroll
  for (int h = 0; h < HH; ++h) {
    float ex = __expf(alpha[(size_t)e * HH + h] - decf(amax[d * HH + h]));
    alpha[(size_t)e * HH + h] = ex;
    atomicAdd(&denom[d * HH + h], ex);
  }
}

// edge pass 3: agg[d,:] += (ex/denom[d]) * xs[s,:]
// 128 consecutive lanes share one edge -> coalesced gather + coalesced atomics,
// all hitting L2 (xs, agg each 25.6MB << 192MB L2).
__global__ void edge_scatter_kernel(const int* __restrict__ src, const int* __restrict__ dst,
                                    const float* __restrict__ alpha, const float* __restrict__ denom,
                                    const float* __restrict__ xs, float* __restrict__ agg) {
  const size_t total = (size_t)ET * C2;
  for (size_t i = (size_t)blockIdx.x * blockDim.x + threadIdx.x; i < total;
       i += (size_t)gridDim.x * blockDim.x) {
    int c = (int)(i & (C2 - 1));
    int e = (int)(i >> 7);
    int s, d;
    if (e < EE) { s = src[e]; d = dst[e]; } else { s = d = e - EE; }
    int h = c >> 6;
    float w = alpha[(size_t)e * HH + h] / (denom[d * HH + h] + 1e-16f);
    atomicAdd(&agg[(size_t)d * C2 + c], w * xs[(size_t)s * C2 + c]);
  }
}

// readout: out[G,HID] += x segment-summed by batch id
__global__ void readout_kernel(const float* __restrict__ x, const int* __restrict__ batch,
                               float* __restrict__ ro) {
  int i = blockIdx.x * blockDim.x + threadIdx.x;
  if (i >= NN * HID) return;
  int n = i >> 6, c = i & 63;
  atomicAdd(&ro[(size_t)batch[n] * HID + c], x[i]);
}

// ---------------------------------------------------------------------------
// host launcher
// ---------------------------------------------------------------------------
static inline unsigned divup(size_t a, unsigned b) { return (unsigned)((a + b - 1) / b); }

extern "C" void kernel_launch(void* const* d_in, const int* in_sizes, int n_in,
                              void* d_out, int out_size, void* d_ws, size_t ws_size,
                              hipStream_t stream) {
  (void)in_sizes; (void)n_in; (void)out_size; (void)ws_size;

  const float* nf      = (const float*)d_in[0];            // [N, 92]
  const int*   ei      = (const int*)d_in[1];              // [2, E]
  const float* aw      = (const float*)d_in[2];            // [N, 1]
  const int*   batch   = (const int*)d_in[3];              // [N]
  const float* embW    = (const float*)d_in[4];            // [64, 92]
  // d_in[5] emb_b: cancels exactly through training-mode BN (mean absorbs it)
  const float* embG    = (const float*)d_in[6];            // [64]
  const float* embBeta = (const float*)d_in[7];            // [64]
  const float* srcW    = (const float*)d_in[8];            // [L,128,64]
  const float* attSrc  = (const float*)d_in[9];            // [L,2,64]
  const float* attDst  = (const float*)d_in[10];           // [L,2,64]
  // d_in[11] bias: cancels exactly through BN (constant per-column shift)
  const float* bnG     = (const float*)d_in[12];           // [L,128]
  const float* bnB     = (const float*)d_in[13];           // [L,128]
  const float* outW    = (const float*)d_in[14];           // [L,64,128]

  const int* srcIdx = ei;
  const int* dstIdx = ei + EE;

  float* x  = (float*)d_out;            // x lives in the output buffer [N,64]
  float* ro = (float*)d_out + (size_t)NN * HID;  // readout [G,64]

  // bump-allocate workspace (256B aligned)
  char* w = (char*)d_ws; size_t off = 0;
  auto alloc = [&](size_t bytes) -> void* {
    void* p = w + off;
    off = (off + bytes + 255) & ~(size_t)255;
    return p;
  };
  __bf16* nfbf   = (__bf16*)alloc((size_t)NN * FPAD * 2);
  __bf16* embWbf = (__bf16*)alloc((size_t)HID * FPAD * 2);
  __bf16* srcWbf = (__bf16*)alloc((size_t)LL * C2 * HID * 2);
  __bf16* outWbf = (__bf16*)alloc((size_t)LL * HID * C2 * 2);
  __bf16* xbf    = (__bf16*)alloc((size_t)NN * HID * 2);
  float*  xs     = (float*)alloc((size_t)NN * C2 * 4);
  float*  agg    = (float*)alloc((size_t)NN * C2 * 4);
  __bf16* aggbf  = (__bf16*)alloc((size_t)NN * C2 * 2);
  float*  as_    = (float*)alloc((size_t)NN * HH * 4);
  float*  ad_    = (float*)alloc((size_t)NN * HH * 4);
  unsigned* amax = (unsigned*)alloc((size_t)NN * HH * 4);
  float*  denom  = (float*)alloc((size_t)NN * HH * 4);
  float*  alpha  = (float*)alloc((size_t)ET * HH * 4);
  float*  bsum   = (float*)alloc(C2 * 4);
  float*  bsq    = (float*)alloc(C2 * 4);
  float*  bscale = (float*)alloc(C2 * 4);
  float*  bshift = (float*)alloc(C2 * 4);

  const unsigned TB = 256;
  const unsigned gemm_blocks = divup(NN / 16, TB / 32);   // 8 waves/block

  // ---- convert inputs/weights to bf16 (zero-pad K 92->96) ----
  cvt_pad_bf16_kernel<<<divup((size_t)NN * FPAD, TB), TB, 0, stream>>>(nf, nfbf, NN, FIN, FPAD);
  cvt_pad_bf16_kernel<<<divup((size_t)HID * FPAD, TB), TB, 0, stream>>>(embW, embWbf, HID, FIN, FPAD);
  cvt_pad_bf16_kernel<<<divup((size_t)LL * C2 * HID, TB), TB, 0, stream>>>(srcW, srcWbf, LL * C2, HID, HID);
  cvt_pad_bf16_kernel<<<divup((size_t)LL * HID * C2, TB), TB, 0, stream>>>(outW, outWbf, LL * HID, C2, C2);

  // ---- embedding: GEMM -> BN stats -> BN+ReLU (writes x f32 + x bf16) ----
  wmma_gemm_kernel<HID, FPAD / 32, 0><<<gemm_blocks, TB, 0, stream>>>(
      nfbf, FPAD, embWbf, FPAD, xs, nullptr, nullptr, NN);
  fill_u32_kernel<<<1, TB, 0, stream>>>((unsigned*)bsum, HID, 0u);
  fill_u32_kernel<<<1, TB, 0, stream>>>((unsigned*)bsq, HID, 0u);
  bn_stats_kernel<<<dim3(HID, 32), TB, 0, stream>>>(xs, NN, HID, bsum, bsq);
  bn_finalize_kernel<<<1, TB, 0, stream>>>(bsum, bsq, embG, embBeta, bscale, bshift, HID, 1.0f / NN);
  bn_apply_kernel<<<divup((size_t)NN * HID, TB), TB, 0, stream>>>(
      xs, bscale, bshift, HID - 1, (size_t)NN * HID, 1, x, xbf);

  // ---- layers ----
  for (int l = 0; l < LL; ++l) {
    // xs = x @ lin_src_W[l].T   [N,64] x [64,128]
    wmma_gemm_kernel<C2, HID / 32, 0><<<gemm_blocks, TB, 0, stream>>>(
        xbf, HID, srcWbf + (size_t)l * C2 * HID, HID, xs, nullptr, nullptr, NN);

    attn_dots_kernel<<<divup((size_t)NN * HH, TB), TB, 0, stream>>>(
        xs, attSrc + (size_t)l * HH * HID, attDst + (size_t)l * HH * HID, as_, ad_, NN);

    // zero segment buffers (amax enc-init 0 < enc(any finite); self-loops ensure hit)
    fill_u32_kernel<<<divup((size_t)NN * HH, TB), TB, 0, stream>>>(amax, (size_t)NN * HH, 0u);
    fill_u32_kernel<<<divup((size_t)NN * HH, TB), TB, 0, stream>>>((unsigned*)denom, (size_t)NN * HH, 0u);
    fill_u32_kernel<<<divup((size_t)NN * C2, TB), TB, 0, stream>>>((unsigned*)agg, (size_t)NN * C2, 0u);

    const unsigned eb = divup(ET, TB);
    edge_alpha_kernel<<<eb, TB, 0, stream>>>(srcIdx, dstIdx, as_, ad_, aw, alpha, amax);
    edge_exp_kernel<<<eb, TB, 0, stream>>>(dstIdx, amax, alpha, denom);
    edge_scatter_kernel<<<8192, TB, 0, stream>>>(srcIdx, dstIdx, alpha, denom, xs, agg);

    // BN over agg (bias folded), write bf16 for final GEMM
    fill_u32_kernel<<<1, TB, 0, stream>>>((unsigned*)bsum, C2, 0u);
    fill_u32_kernel<<<1, TB, 0, stream>>>((unsigned*)bsq, C2, 0u);
    bn_stats_kernel<<<dim3(C2, 32), TB, 0, stream>>>(agg, NN, C2, bsum, bsq);
    bn_finalize_kernel<<<1, TB, 0, stream>>>(bsum, bsq, bnG + (size_t)l * C2, bnB + (size_t)l * C2,
                                             bscale, bshift, C2, 1.0f / NN);
    bn_apply_kernel<<<divup((size_t)NN * C2, TB), TB, 0, stream>>>(
        agg, bscale, bshift, C2 - 1, (size_t)NN * C2, 0, nullptr, aggbf);

    // x = bn(agg) @ lin_out_W[l].T + x   (residual fused in store; refresh xbf)
    wmma_gemm_kernel<HID, C2 / 32, 1><<<gemm_blocks, TB, 0, stream>>>(
        aggbf, C2, outWbf + (size_t)l * HID * C2, C2, x, x, xbf, NN);
  }

  // ---- readout: zero then segment-sum by graph id ----
  fill_u32_kernel<<<divup((size_t)GG * HID, TB), TB, 0, stream>>>((unsigned*)ro, (size_t)GG * HID, 0u);
  readout_kernel<<<divup((size_t)NN * HID, TB), TB, 0, stream>>>(x, batch, ro);
}
